// AlphaModel_7550552506812
// MI455X (gfx1250) — compile-verified, matrix operand order checked by reading, not run on hardware
//
#include <hip/hip_runtime.h>
#include <cstdint>
#include <cstddef>

// Problem constants (match reference)
#define DD     1024
#define CC     2048
#define NN     4096
#define NDEVN  4096
#define NSTEPS 16
#define NWARM  2      // WARMUP_ITERS

#define KSTEP  64     // K elements staged per LDS buffer fill

typedef __attribute__((ext_vector_type(4)))  unsigned int v4u;
typedef __attribute__((ext_vector_type(8)))  int          v8i;
typedef __attribute__((ext_vector_type(4)))  int          v4i;
typedef __attribute__((ext_vector_type(8)))  __bf16       v8bf;
typedef __attribute__((ext_vector_type(16))) __bf16       v16bf;
typedef __attribute__((ext_vector_type(8)))  float        v8f;

__device__ __forceinline__ unsigned short f32_to_bf16_bits(float f) {
  __bf16 h = (__bf16)f;
  return __builtin_bit_cast(unsigned short, h);
}

#if __has_builtin(__builtin_amdgcn_tensor_load_to_lds) && __has_builtin(__builtin_amdgcn_s_wait_tensorcnt)
#define HAVE_TDM 1
#else
#define HAVE_TDM 0
#endif

#if HAVE_TDM
// Build a 2D-tile Tensor DMA descriptor (D#) per CDNA5 ISA ch.8 and issue
// tensor_load_to_lds.  Tile = 128 rows x KSTEP bf16 elements, packed row-major
// in LDS.  6-arg builtin on this toolchain:
//   (uint32x4 g0, int32x8 g1, int32x4 g2, int32x4 g3, int32x8 ext, i32 cpol)
__device__ __forceinline__ void tdm_load_tile(const unsigned short* gsrc,
                                              unsigned int lds_off,
                                              unsigned int rows,
                                              unsigned int row_stride_elems) {
  unsigned long long ga = (unsigned long long)(uintptr_t)gsrc;
  v4u g0;
  g0.x = 1u;                                        // count = 1 (valid descriptor)
  g0.y = lds_off;                                   // lds_addr (bytes)
  g0.z = (unsigned int)(ga & 0xFFFFFFFFu);          // global_addr[31:0]
  g0.w = (unsigned int)((ga >> 32) & 0x01FFFFFFu)   // global_addr[56:32]
       | (2u << 30);                                // type = 2 ("image")
  unsigned long long tdim0 = row_stride_elems;      // tensor_dim0 (elements/row)
  unsigned long long tdim1 = rows;                  // tensor_dim1 (#rows)
  unsigned long long str0  = row_stride_elems;      // tensor_dim0_stride
  // group1: 256 bits assembled as 4x u64 per ISA bit layout
  unsigned long long q0 = (1ull << 16)                        // data_size = 1 (2 bytes)
                        | ((tdim0 & 0xFFFFull) << 48);        // tensor_dim0[15:0]
  unsigned long long q1 = ((tdim0 >> 16) & 0xFFFFull)         // tensor_dim0[31:16]
                        | ((tdim1 & 0xFFFFFFFFull) << 16)     // tensor_dim1
                        | ((unsigned long long)KSTEP << 48);  // tile_dim0 = KSTEP
  unsigned long long q2 = ((unsigned long long)rows & 0xFFFFull)   // tile_dim1
                        | ((str0 & 0xFFFFFFFFull) << 32);          // stride0[31:0]
  unsigned long long q3 = (str0 >> 32) & 0xFFFFull;                // stride0[47:32]
  v8i g1;
  g1[0] = (int)(unsigned int)q0;  g1[1] = (int)(unsigned int)(q0 >> 32);
  g1[2] = (int)(unsigned int)q1;  g1[3] = (int)(unsigned int)(q1 >> 32);
  g1[4] = (int)(unsigned int)q2;  g1[5] = (int)(unsigned int)(q2 >> 32);
  g1[6] = (int)(unsigned int)q3;  g1[7] = (int)(unsigned int)(q3 >> 32);
  v4i gz4 = {0, 0, 0, 0};
  v8i gz8 = {0, 0, 0, 0, 0, 0, 0, 0};
  __builtin_amdgcn_tensor_load_to_lds(g0, g1, gz4, gz4, gz8, 0);
}
#endif

// ---------------------------------------------------------------------------
// Unified bf16 WMMA GEMM:  C[M,N](f32) = A[M,K](bf16) * Bt[N,K](bf16)^T
// Block tile 128x128, 256 threads = 8 waves, wave tile 32(M) x 64(N).
// Double-buffered LDS; TDM prefetch of chunk k+1 overlaps WMMA on chunk k
// (in-order TENSORcnt: wait <=2 after issuing 2 new loads => current buffer
//  complete while next buffer streams in behind the math).
// ---------------------------------------------------------------------------
__global__ __launch_bounds__(256)
void gemm_bf16_wmma(const unsigned short* __restrict__ A,
                    const unsigned short* __restrict__ Bt,
                    float* __restrict__ Cm, int M, int N, int K) {
  __shared__ unsigned short sA[2][128 * KSTEP];   // 2 x 16 KB
  __shared__ unsigned short sB[2][128 * KSTEP];   // 2 x 16 KB

  const int tid  = threadIdx.x;
  const int lane = tid & 31;
  const int wave = tid >> 5;
  const int wm   = (wave & 3) * 32;   // wave M offset inside block tile
  const int wn   = (wave >> 2) * 64;  // wave N offset inside block tile
  const int blockM = blockIdx.y * 128;
  const int blockN = blockIdx.x * 128;

  const int half = lane >> 4;         // 0: lanes 0-15, 1: lanes 16-31
  const int lr   = lane & 15;
  const int offE = half * 8;          // per-lane K sub-chunk offset (elements)

  v8f zero = {0.f, 0.f, 0.f, 0.f, 0.f, 0.f, 0.f, 0.f};
  v8f acc[2][4];
  for (int mi = 0; mi < 2; ++mi)
    for (int ni = 0; ni < 4; ++ni)
      acc[mi][ni] = zero;

#if HAVE_TDM
  // Prologue: prefetch chunk 0 into buffer 0.
  if (wave == 0) {
    tdm_load_tile(A  + (size_t)blockM * K,
                  (unsigned int)(uintptr_t)(void*)&sA[0][0], 128u, (unsigned int)K);
    tdm_load_tile(Bt + (size_t)blockN * K,
                  (unsigned int)(uintptr_t)(void*)&sB[0][0], 128u, (unsigned int)K);
  }
#endif

  int buf = 0;
  for (int kk = 0; kk < K; kk += KSTEP) {
#if HAVE_TDM
    if (wave == 0) {
      if (kk + KSTEP < K) {
        // Prefetch next chunk into the other buffer, then wait only for the
        // current buffer's 2 loads (in-order completion => wait tensorcnt<=2).
        tdm_load_tile(A  + (size_t)blockM * K + kk + KSTEP,
                      (unsigned int)(uintptr_t)(void*)&sA[buf ^ 1][0], 128u,
                      (unsigned int)K);
        tdm_load_tile(Bt + (size_t)blockN * K + kk + KSTEP,
                      (unsigned int)(uintptr_t)(void*)&sB[buf ^ 1][0], 128u,
                      (unsigned int)K);
        __builtin_amdgcn_s_wait_tensorcnt(2);
      } else {
        __builtin_amdgcn_s_wait_tensorcnt(0);
      }
    }
#else
    for (int idx = tid; idx < 128 * (KSTEP / 8); idx += 256) {
      int r  = idx / (KSTEP / 8);
      int c8 = (idx % (KSTEP / 8)) * 8;
      *(v8bf*)&sA[buf][r * KSTEP + c8] =
          *(const v8bf*)&A [(size_t)(blockM + r) * K + kk + c8];
      *(v8bf*)&sB[buf][r * KSTEP + c8] =
          *(const v8bf*)&Bt[(size_t)(blockN + r) * K + kk + c8];
    }
#endif
    __syncthreads();   // current buffer visible to all waves

#pragma unroll
    for (int ks = 0; ks < KSTEP; ks += 32) {
      // A fragments: lane holds 16 bf16 of row (wm+mi*16+lr):
      //   lanes 0-15: K{0-7,16-23}, lanes 16-31: K{8-15,24-31} (of this chunk)
      v16bf afrag[2];
      for (int mi = 0; mi < 2; ++mi) {
        const unsigned short* p = &sA[buf][(wm + mi * 16 + lr) * KSTEP + ks + offE];
        v8bf lo = *(const v8bf*)p;
        v8bf hi = *(const v8bf*)(p + 16);
        for (int e = 0; e < 8; ++e) { afrag[mi][e] = lo[e]; afrag[mi][e + 8] = hi[e]; }
      }
      v16bf bfrag[4];
      for (int ni = 0; ni < 4; ++ni) {
        const unsigned short* p = &sB[buf][(wn + ni * 16 + lr) * KSTEP + ks + offE];
        v8bf lo = *(const v8bf*)p;
        v8bf hi = *(const v8bf*)(p + 16);
        for (int e = 0; e < 8; ++e) { bfrag[ni][e] = lo[e]; bfrag[ni][e + 8] = hi[e]; }
      }
      for (int mi = 0; mi < 2; ++mi)
        for (int ni = 0; ni < 4; ++ni)
          acc[mi][ni] = __builtin_amdgcn_wmma_f32_16x16x32_bf16(
              false, afrag[mi], false, bfrag[ni], (short)0, acc[mi][ni], false, false);
    }

    __syncthreads();   // all waves done reading `buf` before next TDM overwrites it
    buf ^= 1;
  }

  // Epilogue: VGPR r -> M = r + half*8 ; lane lr -> N.
  for (int mi = 0; mi < 2; ++mi)
    for (int ni = 0; ni < 4; ++ni)
      for (int r = 0; r < 8; ++r) {
        int row = blockM + wm + mi * 16 + half * 8 + r;
        int col = blockN + wn + ni * 16 + lr;
        Cm[(size_t)row * N + col] = acc[mi][ni][r];
      }
}

// ---------------------------------------------------------------------------
// Elementwise / conversion kernels
// ---------------------------------------------------------------------------
__global__ void copy_f32_kernel(const float* __restrict__ src, float* __restrict__ dst, int n) {
  int i = blockIdx.x * blockDim.x + threadIdx.x;
  if (i < n) dst[i] = src[i];
}

// xn (N x D, f32) -> xn_bf16 (N x D) and xnT_bf16 (D x N)
__global__ void cvt_x_kernel(const float* __restrict__ x,
                             unsigned short* __restrict__ xb,
                             unsigned short* __restrict__ xTb, int n, int d) {
  int id = blockIdx.x * blockDim.x + threadIdx.x;
  if (id >= n * d) return;
  int i = id / d, dd = id % d;
  unsigned short v = f32_to_bf16_bits(x[id]);
  xb[id] = v;
  xTb[(size_t)dd * n + i] = v;
}

__global__ void cvt_f32_bf16_kernel(const float* __restrict__ x,
                                    unsigned short* __restrict__ xb, int n) {
  int i = blockIdx.x * blockDim.x + threadIdx.x;
  if (i < n) xb[i] = f32_to_bf16_bits(x[i]);
}

// theta (D x C, f32) -> W^T bf16 (C x D row-major, K-contiguous for GEMM B)
__global__ void cvt_wT_kernel(const float* __restrict__ theta,
                              unsigned short* __restrict__ wT) {
  int id = blockIdx.x * blockDim.x + threadIdx.x;
  if (id >= DD * CC) return;
  int c = id / DD, d = id % DD;
  wT[id] = f32_to_bf16_bits(theta[(size_t)d * CC + c]);
}

// ---------------------------------------------------------------------------
// Fused softmax backward: dlogitsT[c, i] = alpha_i * (softmax(logits)_ic - 1{c==y_i})
// One block per row (256 threads, 2048 cols).
// ---------------------------------------------------------------------------
__global__ __launch_bounds__(256)
void softmax_dlogits_kernel(const float* __restrict__ logits,
                            const int* __restrict__ y,
                            const float* __restrict__ alpha_row,
                            unsigned short* __restrict__ dlT) {
  __shared__ float red[256];
  const int row = blockIdx.x;
  const int tid = threadIdx.x;
  const float* lp = logits + (size_t)row * CC;

  float vmax = -3.4e38f;
  for (int c = tid; c < CC; c += 256) vmax = fmaxf(vmax, lp[c]);
  red[tid] = vmax; __syncthreads();
  for (int s = 128; s > 0; s >>= 1) {
    if (tid < s) red[tid] = fmaxf(red[tid], red[tid + s]);
    __syncthreads();
  }
  vmax = red[0]; __syncthreads();

  float sum = 0.f;
  for (int c = tid; c < CC; c += 256) sum += __expf(lp[c] - vmax);
  red[tid] = sum; __syncthreads();
  for (int s = 128; s > 0; s >>= 1) {
    if (tid < s) red[tid] += red[tid + s];
    __syncthreads();
  }
  const float lse = vmax + __logf(red[0]);

  const float a  = alpha_row[row];
  const int   yv = y[row];
  for (int c = tid; c < CC; c += 256) {
    float p = __expf(lp[c] - lse);
    float g = a * (p - (c == yv ? 1.f : 0.f));
    dlT[(size_t)c * NN + row] = f32_to_bf16_bits(g);
  }
}

// Per-row dev CE loss: loss_i = logsumexp(logits_i) - logits_i[y_i]
__global__ __launch_bounds__(256)
void dev_loss_row_kernel(const float* __restrict__ logits,
                         const int* __restrict__ y,
                         float* __restrict__ loss) {
  __shared__ float red[256];
  const int row = blockIdx.x;
  const int tid = threadIdx.x;
  const float* lp = logits + (size_t)row * CC;

  float vmax = -3.4e38f;
  for (int c = tid; c < CC; c += 256) vmax = fmaxf(vmax, lp[c]);
  red[tid] = vmax; __syncthreads();
  for (int s = 128; s > 0; s >>= 1) {
    if (tid < s) red[tid] = fmaxf(red[tid], red[tid + s]);
    __syncthreads();
  }
  vmax = red[0]; __syncthreads();

  float sum = 0.f;
  for (int c = tid; c < CC; c += 256) sum += __expf(lp[c] - vmax);
  red[tid] = sum; __syncthreads();
  for (int s = 128; s > 0; s >>= 1) {
    if (tid < s) red[tid] += red[tid + s];
    __syncthreads();
  }
  if (tid == 0) loss[row] = vmax + __logf(red[0]) - lp[y[row]];
}

// ---------------------------------------------------------------------------
// Deterministic reductions + SGD update
// ---------------------------------------------------------------------------
__global__ __launch_bounds__(256)
void sumsq_partial_kernel(const float* __restrict__ g, int n, float* __restrict__ part) {
  __shared__ float red[256];
  float s = 0.f;
  for (int i = blockIdx.x * 256 + threadIdx.x; i < n; i += gridDim.x * 256) s += g[i] * g[i];
  red[threadIdx.x] = s; __syncthreads();
  for (int st = 128; st > 0; st >>= 1) {
    if (threadIdx.x < st) red[threadIdx.x] += red[threadIdx.x + st];
    __syncthreads();
  }
  if (threadIdx.x == 0) part[blockIdx.x] = red[0];
}

__global__ __launch_bounds__(256)
void finalize_scale_kernel(const float* __restrict__ part, int np,
                           const float* __restrict__ eta, float tfac,
                           float* __restrict__ scale) {
  __shared__ float red[256];
  float s = 0.f;
  for (int i = threadIdx.x; i < np; i += 256) s += part[i];
  red[threadIdx.x] = s; __syncthreads();
  for (int st = 128; st > 0; st >>= 1) {
    if (threadIdx.x < st) red[threadIdx.x] += red[threadIdx.x + st];
    __syncthreads();
  }
  if (threadIdx.x == 0) {
    float gn   = sqrtf(red[0]);
    float coef = fminf(1.0f / (gn + 1e-6f), 1.0f);   // CLIP = 1.0
    *scale = eta[0] * tfac * coef;
  }
}

__global__ void update_theta_kernel(float* __restrict__ theta,
                                    const float* __restrict__ grad,
                                    const float* __restrict__ scale, int n) {
  int i = blockIdx.x * blockDim.x + threadIdx.x;
  if (i < n) theta[i] -= (*scale) * grad[i];
}

__global__ __launch_bounds__(256)
void dev_loss_reduce_kernel(const float* __restrict__ loss, float* __restrict__ out) {
  __shared__ float red[256];
  float s = 0.f;
  for (int i = threadIdx.x; i < NDEVN; i += 256) s += loss[i];
  red[threadIdx.x] = s; __syncthreads();
  for (int st = 128; st > 0; st >>= 1) {
    if (threadIdx.x < st) red[threadIdx.x] += red[threadIdx.x + st];
    __syncthreads();
  }
  if (threadIdx.x == 0) *out = red[0] / (float)NDEVN;
}

__global__ void area_kernel(float* __restrict__ out) {
  if (threadIdx.x == 0 && blockIdx.x == 0) {
    float s = 0.f;
    for (int t = 0; t < NSTEPS; ++t) s += out[1 + t];
    out[0] = s / (float)NSTEPS;
  }
}

// ---------------------------------------------------------------------------
// Launch orchestration
// ---------------------------------------------------------------------------
extern "C" void kernel_launch(void* const* d_in, const int* in_sizes, int n_in,
                              void* d_out, int out_size, void* d_ws, size_t ws_size,
                              hipStream_t stream) {
  (void)in_sizes; (void)n_in; (void)out_size; (void)ws_size;
  const float* theta_in = (const float*)d_in[0];
  const float* alpha    = (const float*)d_in[1];   // [NSTEPS, NN]
  const float* xn       = (const float*)d_in[2];   // [NN, DD]
  const int*   yn       = (const int*)d_in[3];
  const float* dev_xn   = (const float*)d_in[4];   // [NDEVN, DD]
  const int*   dev_yn   = (const int*)d_in[5];
  const float* eta      = (const float*)d_in[6];   // scalar
  float* out = (float*)d_out;                      // [0]=area, [1..16]=dev losses

  // Workspace carve-up (256B aligned)
  char* p = (char*)d_ws;
  auto take = [&](size_t bytes) { char* r = p; p += (bytes + 255) & ~(size_t)255; return r; };
  float*          theta_cur = (float*)         take((size_t)DD * CC * 4);
  unsigned short* xnb       = (unsigned short*)take((size_t)NN * DD * 2);
  unsigned short* xnTb      = (unsigned short*)take((size_t)DD * NN * 2);
  unsigned short* devb      = (unsigned short*)take((size_t)NDEVN * DD * 2);
  unsigned short* wTb       = (unsigned short*)take((size_t)CC * DD * 2);
  float*          logits    = (float*)         take((size_t)NN * CC * 4);  // reused for dev
  unsigned short* dlTb      = (unsigned short*)take((size_t)CC * NN * 2);
  float*          grad      = (float*)         take((size_t)DD * CC * 4);
  float*          part      = (float*)         take(512 * 4);
  float*          scale     = (float*)         take(4);
  float*          devloss   = (float*)         take((size_t)NDEVN * 4);

  const int nTheta = DD * CC;                 // 2M
  const int nX     = NN * DD;                 // 4M

  copy_f32_kernel<<<(nTheta + 255) / 256, 256, 0, stream>>>(theta_in, theta_cur, nTheta);
  cvt_x_kernel<<<(nX + 255) / 256, 256, 0, stream>>>(xn, xnb, xnTb, NN, DD);
  cvt_f32_bf16_kernel<<<(nX + 255) / 256, 256, 0, stream>>>(dev_xn, devb, nX);
  cvt_wT_kernel<<<(nTheta + 255) / 256, 256, 0, stream>>>(theta_cur, wTb);

  for (int t = 0; t < NSTEPS; ++t) {
    const float tfac = (t < NWARM) ? (float)t / (float)NWARM : 1.0f;

    // logits = xn @ W  (M=NN, N=CC, K=DD)
    gemm_bf16_wmma<<<dim3(CC / 128, NN / 128), 256, 0, stream>>>(
        xnb, wTb, logits, NN, CC, DD);
    // dlogitsT = [alpha * (softmax - onehot)]^T  (bf16)
    softmax_dlogits_kernel<<<NN, 256, 0, stream>>>(logits, yn, alpha + (size_t)t * NN, dlTb);
    // grad = xn^T @ dlogits  (M=DD, N=CC, K=NN)
    gemm_bf16_wmma<<<dim3(CC / 128, DD / 128), 256, 0, stream>>>(
        xnTb, dlTb, grad, DD, CC, NN);
    // clip-by-global-norm scale, then theta -= scale * grad
    sumsq_partial_kernel<<<512, 256, 0, stream>>>(grad, nTheta, part);
    finalize_scale_kernel<<<1, 256, 0, stream>>>(part, 512, eta, tfac, scale);
    update_theta_kernel<<<(nTheta + 255) / 256, 256, 0, stream>>>(theta_cur, grad, scale, nTheta);
    // refresh bf16 W^T for dev pass / next step
    cvt_wT_kernel<<<(nTheta + 255) / 256, 256, 0, stream>>>(theta_cur, wTb);
    // dev_logits = dev_xn @ W_new ; dev loss mean -> out[1+t]
    gemm_bf16_wmma<<<dim3(CC / 128, NDEVN / 128), 256, 0, stream>>>(
        devb, wTb, logits, NDEVN, CC, DD);
    dev_loss_row_kernel<<<NDEVN, 256, 0, stream>>>(logits, dev_yn, devloss);
    dev_loss_reduce_kernel<<<1, 256, 0, stream>>>(devloss, out + 1 + t);
  }

  area_kernel<<<1, 32, 0, stream>>>(out);
}
